// DecoderBlock_21526376088288
// MI455X (gfx1250) — compile-verified, matrix operand order checked by reading, not run on hardware
//
#include <hip/hip_runtime.h>

#define B_    16
#define L_    1024
#define E_    512
#define EXP_  2048
#define NTOK  (B_ * L_)
#define NE    ((long)NTOK * E_)
#define KSZ   25
#define PAD_  12
#define TOPK  6
#define EPS_  1e-5f
#define NEG_INF -1e30f

// Async global->LDS path via inline CDNA5 asm (probe-verified on this
// toolchain: assembler accepts global_load_async_to_lds_b128 / s_wait_asynccnt).
#define USE_ASYNC 1

typedef unsigned short u16;
typedef __attribute__((ext_vector_type(16))) __bf16 v16bf;
typedef __attribute__((ext_vector_type(8)))  float  v8f;

// ---------------- helpers ----------------

__device__ __forceinline__ u16 f2bf(float f) {
  union { float f; unsigned u; } c; c.f = f;
  unsigned u = c.u;
  u += 0x7FFFu + ((u >> 16) & 1u);   // round-to-nearest-even
  return (u16)(u >> 16);
}

union Frag16 { v16bf v; uint4 q[2]; };

// CDNA5 16-bit operand layout (ISA 7.12.2): lanes 0-15 -> row = lane,
// K chunks [0..7] and [16..23]; lanes 16-31 -> row = lane-16,
// K chunks [8..15] and [24..31]. Applied to a 32-wide K tile in LDS.
__device__ __forceinline__ v16bf lds_frag16(const u16* s, int ld, int row0) {
  int lane = threadIdx.x & 31;
  int r    = row0 + (lane & 15);
  int kofs = (lane & 16) >> 1;                  // +0 or +8
  const u16* p = s + r * ld + kofs;
  Frag16 f;
  f.q[0] = *reinterpret_cast<const uint4*>(p);        // K chunk lo (8 bf16)
  f.q[1] = *reinterpret_cast<const uint4*>(p + 16);   // K chunk hi
  return f.v;
}

#if USE_ASYNC
// GLOBAL_LOAD_ASYNC_TO_LDS_B128 (VGLOBAL op 98): VDST = LDS byte address,
// VADDR = 64-bit global address. Generic shared pointers carry the LDS
// offset in ADDR[31:0] (flat aperture rules, ISA 10.2), so a 32-bit
// truncation yields the DS address.
__device__ __forceinline__ void async_cp16(const u16* g, u16* l) {
  unsigned lo = (unsigned)(unsigned long long)l;
  asm volatile("global_load_async_to_lds_b128 %0, %1, off"
               :: "v"(lo), "v"(g) : "memory");
}
__device__ __forceinline__ void async_wait0() {
  asm volatile("s_wait_asynccnt 0x0" ::: "memory");
}
#endif

// C/D 16x16 f32 layout: VGPR i -> M = i (lanes 0-15) or 8+i (lanes 16-31), N = lane&15
template <int GELU>
__device__ __forceinline__ void store_tile(float* Cb, int ldc, int row0, int col0,
                                           v8f acc, const float* bias) {
  int lane = threadIdx.x & 31;
  int n  = col0 + (lane & 15);
  int m0 = row0 + ((lane & 16) >> 1);
  float bval = bias ? bias[n] : 0.0f;
#pragma unroll
  for (int i = 0; i < 8; ++i) {
    float v = acc[i] + bval;
    if (GELU) v = 0.5f * v * (1.0f + erff(v * 0.70710678118654752f));
    Cb[(size_t)(m0 + i) * ldc + n] = v;
  }
}

// ---------------- WMMA GEMM: C[M,N] = A[M,K](bf16) * Bt[N,K]^T(bf16) + bias ----------------
// block = 256 threads = 8 waves; block tile 256(M) x 64(N); wave tile 32 x 64;
// K step 32 -> 8 WMMAs per wave per step.  A/B K-tiles double-buffered in LDS
// (row stride 40 -> 80B, 16B aligned, bank-rotating so ds_load_b128 fragment
// reads stay conflict-free).  Next tile streamed via ASYNCcnt-tracked
// global_load_async_to_lds_b128 overlapping the WMMAs.
#define LDA 40
#define TM  256

template <int GELU>
__global__ __launch_bounds__(256)
void gemm_bf16(const u16* __restrict__ A, const u16* __restrict__ Bt,
               float* __restrict__ C, const float* __restrict__ bias,
               int M, int N, int K, long strA, long strB, long strC) {
  __shared__ u16 shA[2][TM * LDA];
  __shared__ u16 shB[2][64 * LDA];
  const u16* Ab = A  + (size_t)blockIdx.z * strA;
  const u16* Bb = Bt + (size_t)blockIdx.z * strB;
  float*     Cb = C  + (size_t)blockIdx.z * strC;
  const int tid    = threadIdx.x;
  const int wave   = tid >> 5;
  const int rowBlk = blockIdx.y * TM;
  const int colBlk = blockIdx.x * 64;
  // staging indices: each thread moves five 16B segments per K step
  const int sr  = tid >> 2;          // 0..63
  const int ssg = (tid & 3) * 8;     // bf16 offset within 32-wide K tile

  // ---- prologue: stage K tile 0 into buffer 0 ----
#if USE_ASYNC
#pragma unroll
  for (int r4 = 0; r4 < 4; ++r4)
    async_cp16(Ab + (size_t)(rowBlk + sr + 64 * r4) * K + ssg,
               &shA[0][(sr + 64 * r4) * LDA + ssg]);
  async_cp16(Bb + (size_t)(colBlk + sr) * K + ssg, &shB[0][sr * LDA + ssg]);
  async_wait0();
#else
  {
    uint4 a[4], b0;
#pragma unroll
    for (int r4 = 0; r4 < 4; ++r4)
      a[r4] = *reinterpret_cast<const uint4*>(Ab + (size_t)(rowBlk + sr + 64 * r4) * K + ssg);
    b0 = *reinterpret_cast<const uint4*>(Bb + (size_t)(colBlk + sr) * K + ssg);
#pragma unroll
    for (int r4 = 0; r4 < 4; ++r4)
      *reinterpret_cast<uint4*>(&shA[0][(sr + 64 * r4) * LDA + ssg]) = a[r4];
    *reinterpret_cast<uint4*>(&shB[0][sr * LDA + ssg]) = b0;
  }
#endif
  __syncthreads();

  v8f acc0 = {}, acc1 = {}, acc2 = {}, acc3 = {};
  v8f acc4 = {}, acc5 = {}, acc6 = {}, acc7 = {};
  int buf = 0;
  for (int k0 = 0; k0 < K; k0 += 32) {
    const int nbuf = buf ^ 1;
    const bool more = (k0 + 32) < K;
#if USE_ASYNC
    if (more) {       // issue next tile; transfers overlap the WMMAs below
#pragma unroll
      for (int r4 = 0; r4 < 4; ++r4)
        async_cp16(Ab + (size_t)(rowBlk + sr + 64 * r4) * K + k0 + 32 + ssg,
                   &shA[nbuf][(sr + 64 * r4) * LDA + ssg]);
      async_cp16(Bb + (size_t)(colBlk + sr) * K + k0 + 32 + ssg,
                 &shB[nbuf][sr * LDA + ssg]);
    }
#else
    uint4 a[4], b0;
    if (more) {       // global loads in flight while WMMAs execute
#pragma unroll
      for (int r4 = 0; r4 < 4; ++r4)
        a[r4] = *reinterpret_cast<const uint4*>(Ab + (size_t)(rowBlk + sr + 64 * r4) * K + k0 + 32 + ssg);
      b0 = *reinterpret_cast<const uint4*>(Bb + (size_t)(colBlk + sr) * K + k0 + 32 + ssg);
    }
#endif
    // compute on current buffer: 2 A frags x 4 B frags = 8 WMMAs
    v16bf fa0 = lds_frag16(shA[buf], LDA, wave * 32);
    v16bf fa1 = lds_frag16(shA[buf], LDA, wave * 32 + 16);
    v16bf f0  = lds_frag16(shB[buf], LDA, 0);
    v16bf f1  = lds_frag16(shB[buf], LDA, 16);
    v16bf f2  = lds_frag16(shB[buf], LDA, 32);
    v16bf f3  = lds_frag16(shB[buf], LDA, 48);
    acc0 = __builtin_amdgcn_wmma_f32_16x16x32_bf16(false, fa0, false, f0, (short)0, acc0, false, false);
    acc1 = __builtin_amdgcn_wmma_f32_16x16x32_bf16(false, fa0, false, f1, (short)0, acc1, false, false);
    acc2 = __builtin_amdgcn_wmma_f32_16x16x32_bf16(false, fa0, false, f2, (short)0, acc2, false, false);
    acc3 = __builtin_amdgcn_wmma_f32_16x16x32_bf16(false, fa0, false, f3, (short)0, acc3, false, false);
    acc4 = __builtin_amdgcn_wmma_f32_16x16x32_bf16(false, fa1, false, f0, (short)0, acc4, false, false);
    acc5 = __builtin_amdgcn_wmma_f32_16x16x32_bf16(false, fa1, false, f1, (short)0, acc5, false, false);
    acc6 = __builtin_amdgcn_wmma_f32_16x16x32_bf16(false, fa1, false, f2, (short)0, acc6, false, false);
    acc7 = __builtin_amdgcn_wmma_f32_16x16x32_bf16(false, fa1, false, f3, (short)0, acc7, false, false);
#if USE_ASYNC
    if (more) async_wait0();
#else
    if (more) {
#pragma unroll
      for (int r4 = 0; r4 < 4; ++r4)
        *reinterpret_cast<uint4*>(&shA[nbuf][(sr + 64 * r4) * LDA + ssg]) = a[r4];
      *reinterpret_cast<uint4*>(&shB[nbuf][sr * LDA + ssg]) = b0;
    }
#endif
    __syncthreads();
    buf = nbuf;
  }
  const int row0 = rowBlk + wave * 32;
  store_tile<GELU>(Cb, N, row0,      colBlk +  0, acc0, bias);
  store_tile<GELU>(Cb, N, row0,      colBlk + 16, acc1, bias);
  store_tile<GELU>(Cb, N, row0,      colBlk + 32, acc2, bias);
  store_tile<GELU>(Cb, N, row0,      colBlk + 48, acc3, bias);
  store_tile<GELU>(Cb, N, row0 + 16, colBlk +  0, acc4, bias);
  store_tile<GELU>(Cb, N, row0 + 16, colBlk + 16, acc5, bias);
  store_tile<GELU>(Cb, N, row0 + 16, colBlk + 32, acc6, bias);
  store_tile<GELU>(Cb, N, row0 + 16, colBlk + 48, acc7, bias);
}

// ---------------- elementwise / data movement ----------------

__global__ void cvt_bf16(const float* __restrict__ in, u16* __restrict__ out, long n) {
  long stride = (long)gridDim.x * blockDim.x;
  for (long i = (long)blockIdx.x * blockDim.x + threadIdx.x; i < n; i += stride)
    out[i] = f2bf(in[i]);
}

// Wt[n,k] = W[k,n] in bf16 (W is K x N row-major fp32)
__global__ void transpose_bf16(const float* __restrict__ W, u16* __restrict__ Wt, int K, int N) {
  long total = (long)K * N;
  long stride = (long)gridDim.x * blockDim.x;
  for (long i = (long)blockIdx.x * blockDim.x + threadIdx.x; i < total; i += stride) {
    long n = i / K, k = i % K;
    Wt[i] = f2bf(W[k * (long)N + n]);
  }
}

// ---------------- autocorrelation pieces ----------------

// mean_corr[b,tau] = (1/E) * sum_t qk[b, (t+tau)%L, t]
__global__ void corr_mean(const float* __restrict__ qk, float* __restrict__ mc) {
  __shared__ float red[256];
  int b = blockIdx.x / L_, tau = blockIdx.x % L_;
  const float* m = qk + (size_t)b * L_ * L_;
  float s = 0.0f;
  for (int t = threadIdx.x; t < L_; t += 256) {
    int row = (t + tau) & (L_ - 1);
    s += m[(size_t)row * L_ + t];
  }
  red[threadIdx.x] = s; __syncthreads();
  for (int o = 128; o > 0; o >>= 1) {
    if (threadIdx.x < (unsigned)o) red[threadIdx.x] += red[threadIdx.x + o];
    __syncthreads();
  }
  if (threadIdx.x == 0) mc[blockIdx.x] = red[0] * (1.0f / E_);
}

__global__ void topk_softmax(const float* __restrict__ mc, float* __restrict__ w, int* __restrict__ d) {
  __shared__ float vals[L_];
  __shared__ float bv[256];
  __shared__ int   bi[256];
  __shared__ float tw[TOPK];
  __shared__ int   td[TOPK];
  int b = blockIdx.x;
  for (int i = threadIdx.x; i < L_; i += 256) vals[i] = mc[b * L_ + i];
  __syncthreads();
  for (int j = 0; j < TOPK; ++j) {
    float best = NEG_INF; int bidx = 0;
    for (int i = threadIdx.x; i < L_; i += 256)
      if (vals[i] > best) { best = vals[i]; bidx = i; }
    bv[threadIdx.x] = best; bi[threadIdx.x] = bidx; __syncthreads();
    for (int o = 128; o > 0; o >>= 1) {
      if (threadIdx.x < (unsigned)o && bv[threadIdx.x + o] > bv[threadIdx.x]) {
        bv[threadIdx.x] = bv[threadIdx.x + o]; bi[threadIdx.x] = bi[threadIdx.x + o];
      }
      __syncthreads();
    }
    if (threadIdx.x == 0) { tw[j] = bv[0]; td[j] = bi[0]; vals[bi[0]] = NEG_INF; }
    __syncthreads();
  }
  if (threadIdx.x == 0) {
    float mx = tw[0], s = 0.0f, e[TOPK];
    for (int j = 0; j < TOPK; ++j) { e[j] = expf(tw[j] - mx); s += e[j]; }
    for (int j = 0; j < TOPK; ++j) { w[b * TOPK + j] = e[j] / s; d[b * TOPK + j] = td[j]; }
  }
}

// out[b,t,e] = sum_j w[b,j] * v[b,(t+d[b,j])%L, e]
__global__ void aggregate(const float* __restrict__ v, const float* __restrict__ w,
                          const int* __restrict__ d, float* __restrict__ out) {
  long idx = (long)blockIdx.x * blockDim.x + threadIdx.x;
  if (idx >= NE) return;
  int  e  = (int)(idx % E_);
  long bt = idx / E_;
  int  t  = (int)(bt % L_);
  long b  = bt / L_;
  float acc = 0.0f;
#pragma unroll
  for (int j = 0; j < TOPK; ++j) {
    int dl = d[b * TOPK + j];
    int tt = (t + dl) & (L_ - 1);
    acc += w[b * TOPK + j] * v[((size_t)b * L_ + tt) * E_ + e];
  }
  out[idx] = acc;
}

// ---------------- series decomposition (25-tap, edge pad) ----------------

__global__ void decomp_add(const float* __restrict__ a, const float* __restrict__ b,
                           float* __restrict__ seasonal, float* __restrict__ trend, int accum) {
  long idx = (long)blockIdx.x * blockDim.x + threadIdx.x;
  if (idx >= NE) return;
  int  e  = (int)(idx % E_);
  long bt = idx / E_;
  int  t  = (int)(bt % L_);
  long bb = bt / L_;
  const float* ap = a + bb * L_ * (long)E_;
  const float* bp = b + bb * L_ * (long)E_;
  float sum = 0.0f;
  for (int u = t - PAD_; u <= t + PAD_; ++u) {
    int uc = u < 0 ? 0 : (u >= L_ ? L_ - 1 : u);
    long o = (long)uc * E_ + e;
    sum += ap[o] + bp[o];
  }
  float mm   = sum * (1.0f / KSZ);
  long  o0   = (long)t * E_ + e;
  float self = ap[o0] + bp[o0];
  seasonal[idx] = self - mm;
  trend[idx]    = accum ? trend[idx] + mm : mm;
}

// ---------------- layernorm + time-mean subtract ----------------

__global__ void layernorm_tok(const float* __restrict__ x, const float* __restrict__ g,
                              const float* __restrict__ be, float* __restrict__ xh) {
  __shared__ float s1[256], s2[256];
  int tok = blockIdx.x;
  const float* xp = x + (size_t)tok * E_;
  float a = 0.0f, b2 = 0.0f;
  for (int i = threadIdx.x; i < E_; i += 256) { float v = xp[i]; a += v; b2 += v * v; }
  s1[threadIdx.x] = a; s2[threadIdx.x] = b2; __syncthreads();
  for (int o = 128; o > 0; o >>= 1) {
    if (threadIdx.x < (unsigned)o) {
      s1[threadIdx.x] += s1[threadIdx.x + o];
      s2[threadIdx.x] += s2[threadIdx.x + o];
    }
    __syncthreads();
  }
  float mu  = s1[0] * (1.0f / E_);
  float var = s2[0] * (1.0f / E_) - mu * mu;
  float inv = rsqrtf(var + EPS_);
  for (int i = threadIdx.x; i < E_; i += 256)
    xh[(size_t)tok * E_ + i] = (xp[i] - mu) * inv * g[i] + be[i];
}

__global__ void time_mean(const float* __restrict__ xh, float* __restrict__ cm) {
  int idx = blockIdx.x * 256 + threadIdx.x;   // over B*E
  if (idx >= B_ * E_) return;
  int b = idx / E_, e = idx % E_;
  float s = 0.0f;
  for (int t = 0; t < L_; ++t) s += xh[((size_t)b * L_ + t) * E_ + e];
  cm[idx] = s * (1.0f / L_);
}

__global__ void sub_mean(const float* __restrict__ xh, const float* __restrict__ cm,
                         float* __restrict__ out) {
  long idx = (long)blockIdx.x * blockDim.x + threadIdx.x;
  if (idx >= NE) return;
  int  e = (int)(idx % E_);
  long b = idx / ((long)L_ * E_);
  out[idx] = xh[idx] - cm[b * E_ + e];
}

// ---------------- trend conv: circular 3-tap, OIH weights (512,512,3) ----------------

#define TT 8
__global__ void trend_conv(const float* __restrict__ tsum, const float* __restrict__ W,
                           float* __restrict__ out) {
  __shared__ float rows[TT + 2][E_];
  int blk = blockIdx.x;               // B * (L/TT)
  int b   = blk / (L_ / TT);
  int t0  = (blk % (L_ / TT)) * TT;
  for (int i = threadIdx.x; i < (TT + 2) * E_; i += 256) {
    int rr = i / E_, e = i % E_;
    int t = (t0 - 1 + rr + L_) & (L_ - 1);     // wrap pad
    rows[rr][e] = tsum[((size_t)b * L_ + t) * E_ + e];
  }
  __syncthreads();
  for (int o = threadIdx.x; o < E_; o += 256) {
    const float* wp = W + (size_t)o * (E_ * 3);
    float acc[TT];
#pragma unroll
    for (int j = 0; j < TT; ++j) acc[j] = 0.0f;
    for (int i = 0; i < E_; ++i) {
      float w0 = wp[3 * i], w1 = wp[3 * i + 1], w2 = wp[3 * i + 2];
#pragma unroll
      for (int j = 0; j < TT; ++j)
        acc[j] += w0 * rows[j][i] + w1 * rows[j + 1][i] + w2 * rows[j + 2][i];
    }
    for (int j = 0; j < TT; ++j)
      out[((size_t)b * L_ + t0 + j) * E_ + o] = acc[j];
  }
}

// ---------------- host orchestration ----------------

struct WSP {
  float *xcur, *xnew, *tsum, *tA, *vbuf, *attn, *qk, *hbuf, *mc, *topw, *cm;
  int* topd;
  u16 *xb, *encb, *qb, *kb, *aggb, *hb, *wT0, *wT1, *wT2, *wT3, *fc1T, *fc2T;
};

static void attention(const u16* xqb, const u16* kvb,
                      const float* wq, const float* wk, const float* wv, const float* wo,
                      const float* bq, const float* bk, const float* bv, const float* bo,
                      const WSP& w, float* attn_out, hipStream_t s) {
  transpose_bf16<<<1024, 256, 0, s>>>(wq, w.wT0, E_, E_);
  transpose_bf16<<<1024, 256, 0, s>>>(wk, w.wT1, E_, E_);
  transpose_bf16<<<1024, 256, 0, s>>>(wv, w.wT2, E_, E_);
  transpose_bf16<<<1024, 256, 0, s>>>(wo, w.wT3, E_, E_);
  dim3 gProj(E_ / 64, NTOK / TM, 1);
  gemm_bf16<0><<<gProj, 256, 0, s>>>(xqb, w.wT0, w.tA, bq, NTOK, E_, E_, 0, 0, 0);
  cvt_bf16<<<4096, 256, 0, s>>>(w.tA, w.qb, NE);
  gemm_bf16<0><<<gProj, 256, 0, s>>>(kvb, w.wT1, w.tA, bk, NTOK, E_, E_, 0, 0, 0);
  cvt_bf16<<<4096, 256, 0, s>>>(w.tA, w.kb, NE);
  gemm_bf16<0><<<gProj, 256, 0, s>>>(kvb, w.wT2, w.vbuf, bv, NTOK, E_, E_, 0, 0, 0);
  dim3 gCorr(L_ / 64, L_ / TM, B_);
  gemm_bf16<0><<<gCorr, 256, 0, s>>>(w.qb, w.kb, w.qk, nullptr, L_, L_, E_,
                                     (long)L_ * E_, (long)L_ * E_, (long)L_ * L_);
  corr_mean<<<B_ * L_, 256, 0, s>>>(w.qk, w.mc);
  topk_softmax<<<B_, 256, 0, s>>>(w.mc, w.topw, w.topd);
  aggregate<<<(int)(NE / 256), 256, 0, s>>>(w.vbuf, w.topw, w.topd, w.tA);
  cvt_bf16<<<4096, 256, 0, s>>>(w.tA, w.aggb, NE);
  gemm_bf16<0><<<gProj, 256, 0, s>>>(w.aggb, w.wT3, attn_out, bo, NTOK, E_, E_, 0, 0, 0);
}

extern "C" void kernel_launch(void* const* d_in, const int* in_sizes, int n_in,
                              void* d_out, int out_size, void* d_ws, size_t ws_size,
                              hipStream_t stream) {
  const float* x     = (const float*)d_in[0];
  const float* enc   = (const float*)d_in[1];
  const float* sa_wq = (const float*)d_in[2];
  const float* sa_wk = (const float*)d_in[3];
  const float* sa_wv = (const float*)d_in[4];
  const float* sa_wo = (const float*)d_in[5];
  const float* sa_bq = (const float*)d_in[6];
  const float* sa_bk = (const float*)d_in[7];
  const float* sa_bv = (const float*)d_in[8];
  const float* sa_bo = (const float*)d_in[9];
  const float* ca_wq = (const float*)d_in[10];
  const float* ca_wk = (const float*)d_in[11];
  const float* ca_wv = (const float*)d_in[12];
  const float* ca_wo = (const float*)d_in[13];
  const float* ca_bq = (const float*)d_in[14];
  const float* ca_bk = (const float*)d_in[15];
  const float* ca_bv = (const float*)d_in[16];
  const float* ca_bo = (const float*)d_in[17];
  const float* fc1_w = (const float*)d_in[18];
  const float* fc1_b = (const float*)d_in[19];
  const float* fc2_w = (const float*)d_in[20];
  const float* fc2_b = (const float*)d_in[21];
  const float* ln_g  = (const float*)d_in[22];
  const float* ln_b  = (const float*)d_in[23];
  const float* tr_w  = (const float*)d_in[24];

  size_t off = 0;
  char* base = (char*)d_ws;
  auto carve = [&](size_t bytes) -> void* {
    void* p = base + off;
    off += (bytes + 255) & ~(size_t)255;
    return p;
  };
  WSP w;
  const size_t NEf = (size_t)NE * 4, NEh = (size_t)NE * 2;
  w.xcur = (float*)carve(NEf);
  w.xnew = (float*)carve(NEf);
  w.tsum = (float*)carve(NEf);
  w.tA   = (float*)carve(NEf);
  w.vbuf = (float*)carve(NEf);
  w.attn = (float*)carve(NEf);
  w.qk   = (float*)carve((size_t)B_ * L_ * L_ * 4);
  w.hbuf = (float*)carve((size_t)NTOK * EXP_ * 4);
  w.xb   = (u16*)carve(NEh);
  w.encb = (u16*)carve(NEh);
  w.qb   = (u16*)carve(NEh);
  w.kb   = (u16*)carve(NEh);
  w.aggb = (u16*)carve(NEh);
  w.hb   = (u16*)carve((size_t)NTOK * EXP_ * 2);
  w.wT0  = (u16*)carve((size_t)E_ * E_ * 2);
  w.wT1  = (u16*)carve((size_t)E_ * E_ * 2);
  w.wT2  = (u16*)carve((size_t)E_ * E_ * 2);
  w.wT3  = (u16*)carve((size_t)E_ * E_ * 2);
  w.fc1T = (u16*)carve((size_t)E_ * EXP_ * 2);
  w.fc2T = (u16*)carve((size_t)E_ * EXP_ * 2);
  w.mc   = (float*)carve((size_t)B_ * L_ * 4);
  w.topw = (float*)carve((size_t)B_ * TOPK * 4);
  w.topd = (int*)carve((size_t)B_ * TOPK * 4);
  w.cm   = (float*)carve((size_t)B_ * E_ * 4);

  const int nblkNE = (int)(NE / 256);

  // Stage 0: bf16 conversions of activations, FFN weight transposes
  cvt_bf16<<<4096, 256, 0, stream>>>(x,   w.xb,   NE);
  cvt_bf16<<<4096, 256, 0, stream>>>(enc, w.encb, NE);
  transpose_bf16<<<4096, 256, 0, stream>>>(fc1_w, w.fc1T, E_, EXP_);
  transpose_bf16<<<4096, 256, 0, stream>>>(fc2_w, w.fc2T, EXP_, E_);

  // Self-attention (autocorrelation) + decomp 1
  attention(w.xb, w.xb, sa_wq, sa_wk, sa_wv, sa_wo, sa_bq, sa_bk, sa_bv, sa_bo,
            w, w.attn, stream);
  decomp_add<<<nblkNE, 256, 0, stream>>>(x, w.attn, w.xcur, w.tsum, 0);
  cvt_bf16<<<4096, 256, 0, stream>>>(w.xcur, w.xb, NE);

  // Cross-attention + decomp 2
  attention(w.xb, w.encb, ca_wq, ca_wk, ca_wv, ca_wo, ca_bq, ca_bk, ca_bv, ca_bo,
            w, w.attn, stream);
  decomp_add<<<nblkNE, 256, 0, stream>>>(w.xcur, w.attn, w.xnew, w.tsum, 1);
  cvt_bf16<<<4096, 256, 0, stream>>>(w.xnew, w.xb, NE);

  // FFN (GELU fused into fc1 epilogue) + decomp 3
  dim3 gF1(EXP_ / 64, NTOK / TM, 1);
  gemm_bf16<1><<<gF1, 256, 0, stream>>>(w.xb, w.fc1T, w.hbuf, fc1_b, NTOK, EXP_, E_, 0, 0, 0);
  cvt_bf16<<<8192, 256, 0, stream>>>(w.hbuf, w.hb, (long)NTOK * EXP_);
  dim3 gF2(E_ / 64, NTOK / TM, 1);
  gemm_bf16<0><<<gF2, 256, 0, stream>>>(w.hb, w.fc2T, w.attn, fc2_b, NTOK, E_, EXP_, 0, 0, 0);
  decomp_add<<<nblkNE, 256, 0, stream>>>(w.xnew, w.attn, w.xcur, w.tsum, 1);

  // Seasonal output: custom layernorm then subtract time-mean
  float* outp = (float*)d_out;
  layernorm_tok<<<NTOK, 256, 0, stream>>>(w.xcur, ln_g, ln_b, w.tA);
  time_mean<<<(B_ * E_ + 255) / 256, 256, 0, stream>>>(w.tA, w.cm);
  sub_mean<<<nblkNE, 256, 0, stream>>>(w.tA, w.cm, outp);

  // Trend output: circular 3-tap conv of accumulated trend
  trend_conv<<<B_ * (L_ / TT), 256, 0, stream>>>(w.tsum, tr_w, outp + NE);
}
